// ICON_54838142435820
// MI455X (gfx1250) — compile-verified, matrix operand order checked by reading, not run on hardware
//
#include <hip/hip_runtime.h>
#include <hip/hip_bf16.h>
#include <math.h>

// Problem constants (match reference)
#define B_   128
#define T_   128
#define DM   1024
#define DH   1024
#define G3   3072   // 3*DH
#define NC   7
#define BH   (B_*DH)          // 131072
#define TD   ((long)B_*T_*DH) // 16777216

typedef _Float16 f16;
typedef f16   v8h  __attribute__((ext_vector_type(8)));
typedef f16   v16h __attribute__((ext_vector_type(16)));
typedef float v8f  __attribute__((ext_vector_type(8)));

// ---------------------------------------------------------------------------
// WMMA helpers.  CDNA5 wave32 16-bit A/B fragment layout (ISA 7.12.2):
// lane L (M/N = L&15) holds K = k+{0..7}, k+{16..23} if L<16,
//                           K = k+{8..15}, k+{24..31} if L>=16.
// Each lane's fragment is two contiguous 16-byte chunks -> two b128 loads.
// ---------------------------------------------------------------------------
__device__ __forceinline__ v16h ldfrag(const f16* p) {
  v8h lo = *reinterpret_cast<const v8h*>(p);
  v8h hi = *reinterpret_cast<const v8h*>(p + 16);
  return __builtin_shufflevector(lo, hi, 0,1,2,3,4,5,6,7,8,9,10,11,12,13,14,15);
}

__device__ __forceinline__ v8f wmma16(v16h a, v16h b, v8f c) {
  return __builtin_amdgcn_wmma_f32_16x16x32_f16(false, a, false, b,
                                                (short)0, c, false, false);
}

__device__ __forceinline__ float sigm(float x) { return 1.0f / (1.0f + expf(-x)); }

// ---------------------------------------------------------------------------
// One phase (x@Wihᵀ or h@Whhᵀ) of a GRU step for a 32x32 output tile at
// (m0, n0).  W is (3072,1024) row-major f16; gates r/z/q at rows 0/1024/2048.
// Per K-chunk: 2 A loads + 6 B loads feed 12 WMMAs (each B reused for two
// M-subtiles) -> 1.33 b128 loads per WMMA instead of 2.3 with 16-wide M.
// acc index = my*2 + ny.
// ---------------------------------------------------------------------------
__device__ __forceinline__ void mm_phase32(
    const f16* __restrict__ X, long sX, const f16* __restrict__ W,
    int m0, int n0, int lane,
    v8f (&r)[4], v8f (&z)[4], v8f (&q)[4])
{
  const int koff = (lane >> 4) << 3;           // 0 or 8 halfs
  const int nc   = lane & 15;
  const f16* a0 = X + (long)(m0 +      nc) * sX + koff;
  const f16* a1 = X + (long)(m0 + 16 + nc) * sX + koff;
  const f16* wp[3][2];
  #pragma unroll
  for (int g = 0; g < 3; ++g) {
    wp[g][0] = W + ((long)(g * 1024 + n0 +      nc) << 10) + koff;
    wp[g][1] = W + ((long)(g * 1024 + n0 + 16 + nc) << 10) + koff;
  }
  for (int k = 0; k < DH; k += 32) {
    v16h A0 = ldfrag(a0 + k);
    v16h A1 = ldfrag(a1 + k);
    #pragma unroll
    for (int ny = 0; ny < 2; ++ny) {
      v16h br = ldfrag(wp[0][ny] + k);
      r[0 + ny] = wmma16(A0, br, r[0 + ny]);
      r[2 + ny] = wmma16(A1, br, r[2 + ny]);
      v16h bz = ldfrag(wp[1][ny] + k);
      z[0 + ny] = wmma16(A0, bz, z[0 + ny]);
      z[2 + ny] = wmma16(A1, bz, z[2 + ny]);
      v16h bq = ldfrag(wp[2][ny] + k);
      q[0 + ny] = wmma16(A0, bq, q[0 + ny]);
      q[2 + ny] = wmma16(A1, bq, q[2 + ny]);
    }
  }
}

// ---------------------------------------------------------------------------
// One time-step of the two masked local GRUs (grid.y = gru index).
// grid.x = 32 blocks x 4 waves -> 128 waves, each a 32x32 tile of the
// 128x1024 h state.  gi/gh fused in one K=2048 accumulation: r/z merge both
// phases; the n gate keeps inn (i) and hn (qq) separate since r multiplies
// only the hh part.
// ---------------------------------------------------------------------------
__global__ __launch_bounds__(128) void k_gru_step_local(
    const f16* __restrict__ x0,  const f16* __restrict__ x1,
    const f16* __restrict__ wihA, const f16* __restrict__ whhA,
    const f16* __restrict__ wihB, const f16* __restrict__ whhB,
    const float* __restrict__ bihA, const float* __restrict__ bhhA,
    const float* __restrict__ bihB, const float* __restrict__ bhhB,
    const float* __restrict__ mk0, const float* __restrict__ mk1,
    const float* __restrict__ hpA, const float* __restrict__ hpB,
    float* __restrict__ hnA, float* __restrict__ hnB,
    const f16* __restrict__ hp16A, const f16* __restrict__ hp16B,
    f16* __restrict__ hn16A, f16* __restrict__ hn16B,
    f16* __restrict__ seqA, f16* __restrict__ seqB, int t)
{
  const int g    = blockIdx.y;
  const int wid  = blockIdx.x * 4 + (threadIdx.x >> 5);   // 0..127
  const int lane = threadIdx.x & 31;
  const int m0   = (wid >> 5) << 5;                       // 0,32,64,96
  const int n0   = (wid & 31) << 5;                       // 0..992

  const f16*  x    = (g ? x1   : x0)  + (long)t * DM;     // row b stride T*DM
  const f16*  wih  = g ? wihB  : wihA;
  const f16*  whh  = g ? whhB  : whhA;
  const float*bih  = g ? bihB  : bihA;
  const float*bhh  = g ? bhhB  : bhhA;
  const float*mask = g ? mk1   : mk0;
  const float*hp   = g ? hpB   : hpA;
  float*      hn   = g ? hnB   : hnA;
  const f16*  hp16 = g ? hp16B : hp16A;
  f16*        hn16 = g ? hn16B : hn16A;
  f16*        seq  = g ? seqB  : seqA;

  v8f r[4] = {}, z[4] = {}, i[4] = {}, qq[4] = {};
  mm_phase32(x,    (long)T_ * DM, wih, m0, n0, lane, r, z, i);
  mm_phase32(hp16, (long)DH,      whh, m0, n0, lane, r, z, qq);

  #pragma unroll
  for (int my = 0; my < 2; ++my) {
    #pragma unroll
    for (int s = 0; s < 2; ++s) {
      const int col = n0 + s * 16 + (lane & 15);
      const float br = bih[col]        + bhh[col];
      const float bz = bih[1024 + col] + bhh[1024 + col];
      const float bi = bih[2048 + col];
      const float bq = bhh[2048 + col];
      const v8f& R = r[my * 2 + s];  const v8f& Z = z[my * 2 + s];
      const v8f& I = i[my * 2 + s];  const v8f& Q = qq[my * 2 + s];
      #pragma unroll
      for (int e = 0; e < 8; ++e) {
        const int row = m0 + my * 16 + e + ((lane >> 4) << 3);
        const float rv = sigm(R[e] + br);
        const float zv = sigm(Z[e] + bz);
        const float nv = tanhf(I[e] + bi + rv * (Q[e] + bq));
        const float ho = hp[row * DH + col];
        const float nh = (1.0f - zv) * nv + zv * ho;
        const float m  = mask[row * T_ + t];
        const float hc = nh * m + ho * (1.0f - m);
        hn[row * DH + col]   = hc;
        hn16[row * DH + col] = (f16)hc;
        seq[((long)row * T_ + t) * DH + col] = (f16)(hc * m); // own_hidden = h*mask
      }
    }
  }
}

// One step (batch index sb) of a hop GRU: recurrence over axis 0 = batch,
// rows are t (faithful to the reference's seq-first GRU on batch-first data).
__global__ __launch_bounds__(128) void k_gru_step_hop(
    const f16* __restrict__ rin,  const f16* __restrict__ wih,
    const f16* __restrict__ whh,
    const float* __restrict__ bih, const float* __restrict__ bhh,
    const float* __restrict__ mko, const float* __restrict__ mkt,
    const float* __restrict__ hp, float* __restrict__ hn,
    const f16* __restrict__ hp16, f16* __restrict__ hn16,
    float* __restrict__ rout32, f16* __restrict__ rnext16, int sb)
{
  const int wid  = blockIdx.x * 4 + (threadIdx.x >> 5);   // 0..127
  const int lane = threadIdx.x & 31;
  const int m0   = (wid >> 5) << 5;   // row tile over t
  const int n0   = (wid & 31) << 5;

  const f16* x = rin + (long)sb * T_ * DH;   // rows t, stride DH

  v8f r[4] = {}, z[4] = {}, i[4] = {}, qq[4] = {};
  mm_phase32(x,    (long)DH, wih, m0, n0, lane, r, z, i);
  mm_phase32(hp16, (long)DH, whh, m0, n0, lane, r, z, qq);

  #pragma unroll
  for (int my = 0; my < 2; ++my) {
    #pragma unroll
    for (int s = 0; s < 2; ++s) {
      const int col = n0 + s * 16 + (lane & 15);
      const float br = bih[col]        + bhh[col];
      const float bz = bih[1024 + col] + bhh[1024 + col];
      const float bi = bih[2048 + col];
      const float bq = bhh[2048 + col];
      const v8f& R = r[my * 2 + s];  const v8f& Z = z[my * 2 + s];
      const v8f& I = i[my * 2 + s];  const v8f& Q = qq[my * 2 + s];
      #pragma unroll
      for (int e = 0; e < 8; ++e) {
        const int row = m0 + my * 16 + e + ((lane >> 4) << 3);   // row == t
        const float rv = sigm(R[e] + br);
        const float zv = sigm(Z[e] + bz);
        const float nv = tanhf(I[e] + bi + rv * (Q[e] + bq));
        const float ho = hp[row * DH + col];
        const float hc = (1.0f - zv) * nv + zv * ho;
        hn[row * DH + col]   = hc;
        hn16[row * DH + col] = (f16)hc;
        const float am = mko[sb * T_ + row] + mkt[sb * T_ + row];
        const float mv = hc * am;                          // masked rnn_outputs
        const long  oi = ((long)sb * T_ + row) * DH + col;
        rout32[oi]  = mv;
        rnext16[oi] = (f16)mv;                             // next hop's input
      }
    }
  }
}

// ------------------------- small VALU kernels ------------------------------
__global__ void k_cvt(const float* __restrict__ s, f16* __restrict__ d, long n) {
  long i  = (long)blockIdx.x * blockDim.x + threadIdx.x;
  long st = (long)gridDim.x * blockDim.x;
  for (; i < n; i += st) d[i] = (f16)s[i];
}

__global__ void k_zero(unsigned int* __restrict__ p, long n) {
  long i  = (long)blockIdx.x * blockDim.x + threadIdx.x;
  long st = (long)gridDim.x * blockDim.x;
  for (; i < n; i += st) p[i] = 0u;
}

__global__ void k_allhist(const f16* __restrict__ a, const f16* __restrict__ b,
                          f16* __restrict__ o, long n) {
  long i  = (long)blockIdx.x * blockDim.x + threadIdx.x;
  long st = (long)gridDim.x * blockDim.x;
  for (; i < n; i += st) o[i] = (f16)tanhf((float)a[i] + (float)b[i]);
}

__global__ __launch_bounds__(256) void k_iproj(
    const float* __restrict__ bin, const float* __restrict__ win,
    float* __restrict__ iproj) {
  const int idx = blockIdx.x * 256 + threadIdx.x;   // B*DH threads
  const int b = idx >> 10, d = idx & 1023;
  const float4* xr = (const float4*)(bin + (long)b * DM);
  const float4* wr = (const float4*)(win + (long)d * DM);
  float acc = 0.0f;
  for (int i = 0; i < DM / 4; ++i) {
    float4 a = xr[i], w = wr[i];
    acc += a.x * w.x + a.y * w.y + a.z * w.z + a.w * w.w;
  }
  iproj[idx] = tanhf(acc);
}

__global__ __launch_bounds__(128) void k_att(
    const float* __restrict__ rout, const float* __restrict__ iproj,
    const float* __restrict__ mko,  const float* __restrict__ mkt,
    float* __restrict__ att) {
  __shared__ float xp[DH];
  __shared__ float red[T_];
  const int b = blockIdx.x, t = threadIdx.x;
  for (int i = t; i < DH; i += T_) xp[i] = iproj[(long)b * DH + i];
  __syncthreads();
  const float4* r = (const float4*)(rout + ((long)b * T_ + t) * DH);
  float acc = 0.0f;
  for (int i = 0; i < DH / 4; ++i) {
    float4 v = r[i];
    acc += v.x * xp[4 * i] + v.y * xp[4 * i + 1] + v.z * xp[4 * i + 2] + v.w * xp[4 * i + 3];
  }
  const float am  = mko[b * T_ + t] + mkt[b * T_ + t];
  const float pre = am * tanhf(acc) + (1.0f - am) * (-10000.0f);
  red[t] = pre; __syncthreads();
  for (int o = T_ / 2; o > 0; o >>= 1) {
    if (t < o) red[t] = fmaxf(red[t], red[t + o]);
    __syncthreads();
  }
  const float mx = red[0]; __syncthreads();
  const float e = expf(pre - mx);
  red[t] = e; __syncthreads();
  for (int o = T_ / 2; o > 0; o >>= 1) {
    if (t < o) red[t] += red[t + o];
    __syncthreads();
  }
  att[b * T_ + t] = am * (e / red[0]);
}

__global__ __launch_bounds__(256) void k_weighted(
    const float* __restrict__ rout, const float* __restrict__ att,
    float* __restrict__ iproj) {
  __shared__ float sa[T_];
  const int b = blockIdx.y;
  const int d = blockIdx.x * 256 + threadIdx.x;
  if (threadIdx.x < T_) sa[threadIdx.x] = att[b * T_ + threadIdx.x];
  __syncthreads();
  const float* r = rout + (long)b * T_ * DH + d;
  float acc = 0.0f;
  #pragma unroll 4
  for (int t = 0; t < T_; ++t) acc += sa[t] * r[(long)t * DH];
  const float v = iproj[(long)b * DH + d];
  iproj[(long)b * DH + d] = tanhf(v + acc);
}

__global__ __launch_bounds__(128) void k_final(
    const float* __restrict__ iproj, const float* __restrict__ wout,
    const float* __restrict__ bout, float* __restrict__ out) {
  __shared__ float red[128];
  __shared__ float lg[NC];
  const int b = blockIdx.x, tid = threadIdx.x;
  for (int c = 0; c < NC; ++c) {
    float a = 0.0f;
    for (int d = tid; d < DH; d += 128)
      a += iproj[(long)b * DH + d] * wout[(long)c * DH + d];
    red[tid] = a; __syncthreads();
    for (int o = 64; o > 0; o >>= 1) {
      if (tid < o) red[tid] += red[tid + o];
      __syncthreads();
    }
    if (tid == 0) lg[c] = red[0] + bout[c];
    __syncthreads();
  }
  if (tid == 0) {
    float mx = lg[0];
    for (int c = 1; c < NC; ++c) mx = fmaxf(mx, lg[c]);
    float s = 0.0f;
    for (int c = 0; c < NC; ++c) s += expf(lg[c] - mx);
    const float l = logf(s);
    for (int c = 0; c < NC; ++c) out[b * NC + c] = lg[c] - mx - l;
  }
}

// ---------------------------------------------------------------------------
// Host orchestration.  Workspace need ~325 MB (weights f16 50MB, hist f16
// 67MB, hidden seqs f16 67MB, rnnin ping-pong f16 67MB, rnnout f32 67MB,
// states/misc ~6MB).  All launches on `stream`; graph-capture safe.
// ---------------------------------------------------------------------------
extern "C" void kernel_launch(void* const* d_in, const int* in_sizes, int n_in,
                              void* d_out, int out_size, void* d_ws, size_t ws_size,
                              hipStream_t stream) {
  (void)in_sizes; (void)n_in; (void)out_size; (void)ws_size;
  const float* batch_in = (const float*)d_in[0];
  const float* hist_own = (const float*)d_in[1];
  const float* hist_oth = (const float*)d_in[2];
  const float* mask_own = (const float*)d_in[3];
  const float* mask_oth = (const float*)d_in[4];
  const float* wf[8] = { (const float*)d_in[5],  (const float*)d_in[6],   // own ih,hh
                         (const float*)d_in[9],  (const float*)d_in[10],  // oth
                         (const float*)d_in[13], (const float*)d_in[14],  // glob
                         (const float*)d_in[17], (const float*)d_in[18] };// mem
  const float* bih_own = (const float*)d_in[7];  const float* bhh_own = (const float*)d_in[8];
  const float* bih_oth = (const float*)d_in[11]; const float* bhh_oth = (const float*)d_in[12];
  const float* bih_glb = (const float*)d_in[15]; const float* bhh_glb = (const float*)d_in[16];
  const float* bih_mem = (const float*)d_in[19]; const float* bhh_mem = (const float*)d_in[20];
  const float* W_in  = (const float*)d_in[21];
  const float* W_out = (const float*)d_in[22];
  const float* b_out = (const float*)d_in[23];

  char* wsp = (char*)d_ws;
  auto alloc = [&](size_t bytes) -> char* {
    char* p = wsp; wsp += (bytes + 255) & ~(size_t)255; return p;
  };

  f16* w16[8];
  for (int i = 0; i < 8; ++i) w16[i] = (f16*)alloc((size_t)G3 * DH * 2);
  f16*   xo16    = (f16*)alloc(TD * 2);
  f16*   xt16    = (f16*)alloc(TD * 2);
  f16*   seq_o   = (f16*)alloc(TD * 2);
  f16*   seq_t   = (f16*)alloc(TD * 2);
  f16*   rnnin[2]; rnnin[0] = (f16*)alloc(TD * 2); rnnin[1] = (f16*)alloc(TD * 2);
  float* rout32  = (float*)alloc(TD * 4);
  float* hl32    = (float*)alloc((size_t)4 * BH * 4);  // [phase][gru]
  f16*   hl16    = (f16*)  alloc((size_t)4 * BH * 2);
  float* hh32    = (float*)alloc((size_t)2 * BH * 4);  // [phase]
  f16*   hh16    = (f16*)  alloc((size_t)2 * BH * 2);
  float* iproj   = (float*)alloc((size_t)BH * 4);
  float* att     = (float*)alloc((size_t)B_ * T_ * 4);

  // 1) f32 -> f16 conversions (weights + histories)
  for (int i = 0; i < 8; ++i)
    k_cvt<<<2048, 256, 0, stream>>>(wf[i], w16[i], (long)G3 * DH);
  k_cvt<<<2048, 256, 0, stream>>>(hist_own, xo16, TD);
  k_cvt<<<2048, 256, 0, stream>>>(hist_oth, xt16, TD);

  // 2) input projection (independent)
  k_iproj<<<(B_ * DH) / 256, 256, 0, stream>>>(batch_in, W_in, iproj);

  // 3) zero initial local-GRU states (phase 0, both GRUs)
  k_zero<<<512, 256, 0, stream>>>((unsigned int*)hl32, (long)2 * BH);
  k_zero<<<512, 256, 0, stream>>>((unsigned int*)hl16, (long)BH);

  // 4) masked local GRUs over T (both in one launch via grid.y)
  for (int t = 0; t < T_; ++t) {
    const int p = t & 1, q = p ^ 1;
    k_gru_step_local<<<dim3(32, 2), 128, 0, stream>>>(
        xo16, xt16, w16[0], w16[1], w16[2], w16[3],
        bih_own, bhh_own, bih_oth, bhh_oth, mask_own, mask_oth,
        hl32 + (p * 2 + 0) * BH, hl32 + (p * 2 + 1) * BH,
        hl32 + (q * 2 + 0) * BH, hl32 + (q * 2 + 1) * BH,
        hl16 + (p * 2 + 0) * BH, hl16 + (p * 2 + 1) * BH,
        hl16 + (q * 2 + 0) * BH, hl16 + (q * 2 + 1) * BH,
        seq_o, seq_t, t);
  }

  // 5) all_history = tanh(own_hidden + other_hidden) -> hop-0 input (f16)
  k_allhist<<<2048, 256, 0, stream>>>(seq_o, seq_t, rnnin[0], TD);

  // 6) hop loop
  for (int hop = 0; hop < 4; ++hop) {
    const f16*   wih = (hop == 0) ? w16[4] : w16[6];
    const f16*   whh = (hop == 0) ? w16[5] : w16[7];
    const float* bih = (hop == 0) ? bih_glb : bih_mem;
    const float* bhh = (hop == 0) ? bhh_glb : bhh_mem;
    f16* rin   = rnnin[hop & 1];
    f16* rnext = rnnin[(hop & 1) ^ 1];

    k_zero<<<512, 256, 0, stream>>>((unsigned int*)hh32, (long)BH);
    k_zero<<<512, 256, 0, stream>>>((unsigned int*)hh16, (long)BH / 2);

    for (int sb = 0; sb < B_; ++sb) {
      const int p = sb & 1, q = p ^ 1;
      k_gru_step_hop<<<32, 128, 0, stream>>>(
          rin, wih, whh, bih, bhh, mask_own, mask_oth,
          hh32 + p * BH, hh32 + q * BH, hh16 + p * BH, hh16 + q * BH,
          rout32, rnext, sb);
    }

    k_att<<<B_, T_, 0, stream>>>(rout32, iproj, mask_own, mask_oth, att);
    k_weighted<<<dim3(DH / 256, B_), 256, 0, stream>>>(rout32, att, iproj);
  }

  // 7) classifier head + log_softmax
  k_final<<<B_, 128, 0, stream>>>(iproj, W_out, b_out, (float*)d_out);
}
